// MultiHeadSelfAttention_4209067950265
// MI455X (gfx1250) — matrix operand order repaired and checked by reading, not compiled, not run on hardware
//
#include <hip/hip_runtime.h>
#include <hip/hip_bf16.h>

typedef unsigned short u16;
typedef unsigned int   u32;

typedef __attribute__((ext_vector_type(16))) __bf16 v16bf;
typedef __attribute__((ext_vector_type(8)))  float  v8f;

union FragA { v16bf v; u32 u[8]; };

// ---------- bf16 helpers (RNE) ----------
__device__ __forceinline__ u16 f2bf(float f) {
  u32 u = __builtin_bit_cast(u32, f);
  u32 r = u + 0x7FFFu + ((u >> 16) & 1u);
  return (u16)(r >> 16);
}
__device__ __forceinline__ float bf2f(u16 h) {
  u32 u = ((u32)h) << 16;
  return __builtin_bit_cast(float, u);
}

// ---------- CDNA5 async global->LDS copy (ASYNCcnt path) ----------
// Copies 32 bytes per lane: two b128 asyncs; offset: applies to BOTH the
// global and LDS addresses per CDNA5 ISA.
__device__ __forceinline__ void async_copy32(void* lds_ptr, const void* gptr) {
  u32 lds_off = (u32)(size_t)lds_ptr;                 // generic->LDS: low 32 bits
  unsigned long long ga = (unsigned long long)(size_t)gptr;
  asm volatile(
      "global_load_async_to_lds_b128 %0, %1, off\n\t"
      "global_load_async_to_lds_b128 %0, %1, off offset:16"
      :: "v"(lds_off), "v"(ga) : "memory");
}
__device__ __forceinline__ void async_wait0() {
  asm volatile("s_wait_asynccnt 0" ::: "memory");
}

// ---------- problem constants ----------
constexpr int Bc = 2, Tc = 4096, Dc = 512, Hc = 8, DKc = 64;
constexpr int Mrows = Bc * Tc;          // 8192

// ---------- fp32 -> bf16 convert ----------
__global__ void cvt_f32_bf16_k(const float* __restrict__ src, u16* __restrict__ dst, int n) {
  int i = blockIdx.x * blockDim.x + threadIdx.x;
  if (i < n) dst[i] = f2bf(src[i]);
}

// ---------- WMMA GEMM: C = A(bf16, MxK) * B(bf16, KxN) + bias ----------
// MODE 0: write fp32 to out.  MODE 1: split N into (q|k|v), heads, write bf16 [B,H,T,DK].
template <int MODE>
__global__ __launch_bounds__(256) void gemm_bf16_k(
    const u16* __restrict__ A, const u16* __restrict__ Bm,
    const float* __restrict__ bias, float* __restrict__ out,
    u16* __restrict__ qd, u16* __restrict__ kd, u16* __restrict__ vd,
    int M, int N, int K) {
  constexpr int BM = 128, BN = 64, BK = 32, PAD = 8;
  __shared__ u16 As[BM][BK + PAD];   // row-major, K contiguous
  __shared__ u16 Bs[BN][BK + PAD];   // transposed: [n][k], K contiguous

  const int tid  = threadIdx.x;
  const int lane = tid & 31;
  const int w    = tid >> 5;
  const int wm   = w >> 1;          // 0..3
  const int wn   = w & 1;           // 0..1
  const int half = lane >> 4;
  const int l16  = lane & 15;

  const int m0 = blockIdx.y * BM;
  const int n0 = blockIdx.x * BN;

  v8f acc[2][2] = {};

  const int arow = tid >> 1, ahalf = tid & 1;   // A loader: 16 elems/thread
  const int bk   = tid >> 3, bg    = tid & 7;   // B loader: 8 elems/thread

  for (int k0 = 0; k0 < K; k0 += BK) {
    __syncthreads();
    // stage A tile via async global->LDS DMA (pure copy, no transpose)
    const u16* ap = A + (size_t)(m0 + arow) * K + k0 + ahalf * 16;
    async_copy32(&As[arow][ahalf * 16], ap);
    // stage B tile transposed (needs VGPR round-trip)
    const u16* bp = Bm + (size_t)(k0 + bk) * N + n0 + bg * 8;
    union { uint4 v; u16 s[8]; } bu; bu.v = *(const uint4*)bp;
#pragma unroll
    for (int i = 0; i < 8; ++i) Bs[bg * 8 + i][bk] = bu.s[i];
    if (k0 + BK < K)
      __builtin_prefetch(Bm + (size_t)(k0 + BK + bk) * N + n0 + bg * 8, 0, 0);
    async_wait0();
    __syncthreads();

    FragA af[2], bfr[2];
#pragma unroll
    for (int mt = 0; mt < 2; ++mt) {
      const u16* src = &As[wm * 32 + mt * 16 + l16][0];
#pragma unroll
      for (int p = 0; p < 4; ++p) af[mt].u[p]     = *(const u32*)&src[half * 8 + 2 * p];
#pragma unroll
      for (int p = 0; p < 4; ++p) af[mt].u[4 + p] = *(const u32*)&src[16 + half * 8 + 2 * p];
    }
#pragma unroll
    for (int nt = 0; nt < 2; ++nt) {
      const u16* src = &Bs[wn * 32 + nt * 16 + l16][0];
#pragma unroll
      for (int p = 0; p < 8; ++p) bfr[nt].u[p] = *(const u32*)&src[(p >> 2) * 16 + half * 8 + 2 * (p & 3)];
    }
#pragma unroll
    for (int mt = 0; mt < 2; ++mt)
#pragma unroll
      for (int nt = 0; nt < 2; ++nt)
        acc[mt][nt] = __builtin_amdgcn_wmma_f32_16x16x32_bf16(
            false, af[mt].v, false, bfr[nt].v, (short)0, acc[mt][nt], false, false);
  }

  // epilogue
#pragma unroll
  for (int mt = 0; mt < 2; ++mt) {
#pragma unroll
    for (int nt = 0; nt < 2; ++nt) {
#pragma unroll
      for (int r = 0; r < 8; ++r) {
        const int gm = m0 + wm * 32 + mt * 16 + half * 8 + r;
        const int gn = n0 + wn * 32 + nt * 16 + l16;
        const float val = acc[mt][nt][r] + bias[gn];
        if (MODE == 0) {
          out[(size_t)gm * N + gn] = val;
        } else {
          const int part = gn >> 9;          // 0:q 1:k 2:v  (D=512)
          const int wi   = gn & 511;
          const int hh   = wi >> 6;
          const int dk   = wi & 63;
          const int bb   = gm >> 12;         // T=4096
          const int tt   = gm & 4095;
          u16* dst = (part == 0) ? qd : (part == 1) ? kd : vd;
          dst[(((size_t)bb * Hc + hh) * Tc + tt) * DKc + dk] = f2bf(val);
        }
      }
    }
  }
}

// ---------- RoPE in-place on Q,K  [B,H,T,DK] bf16 ----------
__global__ void rope_k(u16* __restrict__ qb, u16* __restrict__ kb) {
  const int i = threadIdx.x;            // 0..31 (half of DK)
  const int t = blockIdx.x;
  const int h = blockIdx.y;
  const int b = blockIdx.z;
  const size_t base = (((size_t)b * Hc + h) * Tc + t) * DKc;
  const float theta = __powf(10000.0f, -(float)i / 32.0f);
  float s, c;
  __sincosf((float)t * theta, &s, &c);
  const float q1 = bf2f(qb[base + i]), q2 = bf2f(qb[base + i + 32]);
  qb[base + i]      = f2bf(q1 * c - q2 * s);
  qb[base + i + 32] = f2bf(q1 * s + q2 * c);
  const float k1 = bf2f(kb[base + i]), k2 = bf2f(kb[base + i + 32]);
  kb[base + i]      = f2bf(k1 * c - k2 * s);
  kb[base + i + 32] = f2bf(k1 * s + k2 * c);
}

// ---------- Flash attention: 64-query tile / block, stream 64-key tiles ----------
__global__ __launch_bounds__(256) void flash_attn_k(
    const u16* __restrict__ Q, const u16* __restrict__ Kb,
    const u16* __restrict__ Vb, u16* __restrict__ ctx) {
  constexpr int PADH = 8, SPAD = 4;
  __shared__ u16  qs [64][DKc + PADH];   // [query][dk]
  __shared__ u16  ks [64][DKc + PADH];   // [key][dk]
  __shared__ u16  vts[DKc][64 + PADH];   // transposed: [dk][key]
  __shared__ u16  ps [64][64 + PADH];    // P bf16 [query][key]
  __shared__ float ss[64][64 + SPAD];    // S f32
  __shared__ float rowm[64], rowl[64], rowsc[64];

  const int tid  = threadIdx.x;
  const int lane = tid & 31;
  const int w    = tid >> 5;
  const int half = lane >> 4;
  const int l16  = lane & 15;
  const int sm   = w >> 1;               // query-row tile of this wave (0..3)
  const int wn2  = w & 1;                // column half (0..1)

  const int qt = blockIdx.x, h = blockIdx.y, b = blockIdx.z;
  const size_t headbase = ((size_t)b * Hc + h) * Tc;
  const int q0 = qt * 64;

  const int row = tid >> 2, seg = tid & 3;   // shared staging decomposition

  { // stage Q tile via async DMA
    const u16* src = Q + (headbase + q0 + row) * DKc + seg * 16;
    async_copy32(&qs[row][seg * 16], src);
  }
  if (tid < 64) { rowm[tid] = -3.0e38f; rowl[tid] = 0.0f; }
  async_wait0();
  __syncthreads();

  // preload Q A-fragments for both k-steps of DK=64
  FragA qa[2];
#pragma unroll
  for (int kk = 0; kk < 2; ++kk) {
    const u16* src = &qs[sm * 16 + l16][kk * 32];
#pragma unroll
    for (int p = 0; p < 4; ++p) qa[kk].u[p]     = *(const u32*)&src[half * 8 + 2 * p];
#pragma unroll
    for (int p = 0; p < 4; ++p) qa[kk].u[4 + p] = *(const u32*)&src[16 + half * 8 + 2 * p];
  }

  v8f o[2] = {};
  const float scale = 0.125f;            // 1/sqrt(64)

  for (int j0 = 0; j0 < Tc; j0 += 64) {
    __syncthreads();                     // protect LDS reuse across iterations
    { // stage K via async DMA (natural layout), V via VGPRs (transposed)
      const u16* srck = Kb + (headbase + j0 + row) * DKc + seg * 16;
      async_copy32(&ks[row][seg * 16], srck);
      const u16* srcv = Vb + (headbase + j0 + row) * DKc + seg * 16;
      union { uint4 v; u16 s[8]; } t0, t1;
      t0.v = *(const uint4*)srcv;
      t1.v = *(const uint4*)(srcv + 8);
#pragma unroll
      for (int i = 0; i < 8; ++i) vts[seg * 16 + i][row]     = t0.s[i];
#pragma unroll
      for (int i = 0; i < 8; ++i) vts[seg * 16 + 8 + i][row] = t1.s[i];
    }
    async_wait0();
    __syncthreads();

    // S = Q * K^T  (each wave: one 16-row tile x two 16-col tiles)
    v8f sacc[2] = {};
#pragma unroll
    for (int kk = 0; kk < 2; ++kk) {
#pragma unroll
      for (int nt = 0; nt < 2; ++nt) {
        FragA bfrag;
        const u16* src = &ks[(wn2 * 2 + nt) * 16 + l16][kk * 32];
#pragma unroll
        for (int p = 0; p < 8; ++p) bfrag.u[p] = *(const u32*)&src[(p >> 2) * 16 + half * 8 + 2 * (p & 3)];
        sacc[nt] = __builtin_amdgcn_wmma_f32_16x16x32_bf16(
            false, qa[kk].v, false, bfrag.v, (short)0, sacc[nt], false, false);
      }
    }
#pragma unroll
    for (int nt = 0; nt < 2; ++nt)
#pragma unroll
      for (int r = 0; r < 8; ++r)
        ss[sm * 16 + half * 8 + r][(wn2 * 2 + nt) * 16 + l16] = sacc[nt][r] * scale;
    __syncthreads();

    // online softmax, one thread per query row
    if (tid < 64) {
      const int r = tid;
      const float mo = rowm[r];
      float tmax = -3.0e38f;
#pragma unroll 8
      for (int c = 0; c < 64; ++c) tmax = fmaxf(tmax, ss[r][c]);
      const float mn   = fmaxf(mo, tmax);
      const float corr = __expf(mo - mn);
      float sum = 0.0f;
#pragma unroll 8
      for (int c = 0; c < 64; ++c) {
        const float p = __expf(ss[r][c] - mn);
        ps[r][c] = f2bf(p);
        sum += p;
      }
      rowl[r]  = rowl[r] * corr + sum;
      rowm[r]  = mn;
      rowsc[r] = corr;
    }
    __syncthreads();

    // rescale O, accumulate O += P * V
#pragma unroll
    for (int nt = 0; nt < 2; ++nt)
#pragma unroll
      for (int r = 0; r < 8; ++r)
        o[nt][r] *= rowsc[sm * 16 + half * 8 + r];

#pragma unroll
    for (int kk = 0; kk < 2; ++kk) {
      FragA pa;
      const u16* srcp = &ps[sm * 16 + l16][kk * 32];
#pragma unroll
      for (int p = 0; p < 4; ++p) pa.u[p]     = *(const u32*)&srcp[half * 8 + 2 * p];
#pragma unroll
      for (int p = 0; p < 4; ++p) pa.u[4 + p] = *(const u32*)&srcp[16 + half * 8 + 2 * p];
#pragma unroll
      for (int nt = 0; nt < 2; ++nt) {
        FragA vfrag;
        const u16* srcv = &vts[wn2 * 32 + nt * 16 + l16][kk * 32];
#pragma unroll
        for (int p = 0; p < 8; ++p) vfrag.u[p] = *(const u32*)&srcv[(p >> 2) * 16 + half * 8 + 2 * (p & 3)];
        o[nt] = __builtin_amdgcn_wmma_f32_16x16x32_bf16(
            false, pa.v, false, vfrag.v, (short)0, o[nt], false, false);
      }
    }
  }

  __syncthreads();
  // finalize: O /= l, write merged-head bf16 ctx [B,T,D]
#pragma unroll
  for (int r = 0; r < 8; ++r) {
    const int rowq = sm * 16 + half * 8 + r;
    const float inv = 1.0f / rowl[rowq];
#pragma unroll
    for (int nt = 0; nt < 2; ++nt) {
      const float val = o[nt][r] * inv;
      const int dcol = h * DKc + wn2 * 32 + nt * 16 + l16;
      ctx[((size_t)b * Tc + q0 + rowq) * Dc + dcol] = f2bf(val);
    }
  }
}

// ---------- host launcher ----------
extern "C" void kernel_launch(void* const* d_in, const int* in_sizes, int n_in,
                              void* d_out, int out_size, void* d_ws, size_t ws_size,
                              hipStream_t stream) {
  const float* x    = (const float*)d_in[0];
  const float* Wqkv = (const float*)d_in[1];
  const float* bqkv = (const float*)d_in[2];
  const float* Wout = (const float*)d_in[3];
  const float* bout = (const float*)d_in[4];
  float* out = (float*)d_out;

  // workspace carve-up (bytes, 256-aligned)
  char* ws = (char*)d_ws;
  constexpr size_t SZ_XB   = (size_t)Mrows * Dc * 2;        // 8 MB
  constexpr size_t SZ_WQKV = (size_t)Dc * 3 * Dc * 2;       // 1.5 MB
  constexpr size_t SZ_WOUT = (size_t)Dc * Dc * 2;           // 0.5 MB
  constexpr size_t SZ_HEAD = (size_t)Bc * Hc * Tc * DKc * 2;// 8 MB each
  size_t off = 0;
  u16* xb    = (u16*)(ws + off); off += SZ_XB;
  u16* wqkvb = (u16*)(ws + off); off += SZ_WQKV;
  u16* woutb = (u16*)(ws + off); off += SZ_WOUT;
  u16* qb    = (u16*)(ws + off); off += SZ_HEAD;
  u16* kb    = (u16*)(ws + off); off += SZ_HEAD;
  u16* vb    = (u16*)(ws + off); off += SZ_HEAD;
  u16* ctxb  = (u16*)(ws + off); off += SZ_XB;

  // 1) convert operands to bf16
  {
    int n1 = Mrows * Dc, n2 = Dc * 3 * Dc, n3 = Dc * Dc;
    cvt_f32_bf16_k<<<(n1 + 255) / 256, 256, 0, stream>>>(x, xb, n1);
    cvt_f32_bf16_k<<<(n2 + 255) / 256, 256, 0, stream>>>(Wqkv, wqkvb, n2);
    cvt_f32_bf16_k<<<(n3 + 255) / 256, 256, 0, stream>>>(Wout, woutb, n3);
  }
  // 2) QKV projection -> head-major bf16 q/k/v
  gemm_bf16_k<1><<<dim3((3 * Dc) / 64, Mrows / 128), 256, 0, stream>>>(
      xb, wqkvb, bqkv, nullptr, qb, kb, vb, Mrows, 3 * Dc, Dc);
  // 3) RoPE in place on q, k
  rope_k<<<dim3(Tc, Hc, Bc), 32, 0, stream>>>(qb, kb);
  // 4) flash attention -> ctx bf16 [B,T,D]
  flash_attn_k<<<dim3(Tc / 64, Hc, Bc), 256, 0, stream>>>(qb, kb, vb, ctxb);
  // 5) output projection -> fp32 out
  gemm_bf16_k<0><<<dim3(Dc / 64, Mrows / 128), 256, 0, stream>>>(
      ctxb, woutb, bout, out, nullptr, nullptr, nullptr, Mrows, Dc, Dc);
}
